// GroupedQueryAttention_32358283608305
// MI455X (gfx1250) — matrix operand order repaired and checked by reading, compiled, not run on hardware
//
#include <hip/hip_runtime.h>

#define N_EMBD 2048
#define N_HEADS 16
#define NUM_KV_GROUPS 2
#define HEAD_DIM 128
#define GROUP_SIZE 8
#define BATCH 2
#define SEQ 2048

typedef __attribute__((ext_vector_type(16))) __bf16 v16bf;
typedef __attribute__((ext_vector_type(8)))  float  v8f;
typedef __attribute__((ext_vector_type(4)))  unsigned int u32x4;
typedef __attribute__((ext_vector_type(8)))  int i32x8;
typedef __attribute__((ext_vector_type(4)))  int i32x4;

struct U4x2 { uint4 lo, hi; };
static_assert(sizeof(U4x2) == sizeof(v16bf), "frag size mismatch");

__device__ __forceinline__ v16bf ld_frag(const unsigned short* p0, const unsigned short* p1) {
  U4x2 f;
  f.lo = *reinterpret_cast<const uint4*>(p0);
  f.hi = *reinterpret_cast<const uint4*>(p1);
  return __builtin_bit_cast(v16bf, f);
}

__device__ __forceinline__ v8f wmma_bf16(v16bf a, v16bf b, v8f c) {
  // D = A(16x32 bf16) * B(32x16 bf16) + C(16x16 f32)
  return __builtin_amdgcn_wmma_f32_16x16x32_bf16(false, a, false, b, (short)0, c, false, false);
}

__device__ __forceinline__ unsigned short f2bf(float f) {
  unsigned int u = __builtin_bit_cast(unsigned int, f);
  u += 0x7FFFu + ((u >> 16) & 1u);   // round-to-nearest-even
  return (unsigned short)(u >> 16);
}

#if defined(__has_builtin)
#if __has_builtin(__builtin_amdgcn_tensor_load_to_lds) && __has_builtin(__builtin_amdgcn_s_wait_tensorcnt)
#define USE_TDM 1
#endif
#endif
#ifndef USE_TDM
#define USE_TDM 0
#endif

#if USE_TDM
__device__ __forceinline__ unsigned lds_addr_of(const void* p) {
  return (unsigned)(unsigned long long)(__attribute__((address_space(3))) const void*)p;
}

// 2D tile load via Tensor Data Mover: tile (tile_w x tile_h) of bf16 (2B) from a
// row-major matrix with row length/stride = stride_elems, packed densely into LDS.
// D# layout per CDNA5 ISA.
__device__ __forceinline__ void tdm_load_2d(unsigned lds_addr, const void* gptr,
                                            unsigned tile_w, unsigned tile_h,
                                            unsigned stride_elems, unsigned nrows) {
  unsigned long long ga = (unsigned long long)(uintptr_t)gptr;
  u32x4 g0;
  g0.x = 1u;                                              // count=1, user descriptor
  g0.y = lds_addr;                                        // bits 63:32  lds_addr
  g0.z = (unsigned)(ga & 0xFFFFFFFFu);                    // bits 95:64  global_addr lo
  g0.w = (unsigned)((ga >> 32) & 0x1FFFFFFu) | (2u << 30);// addr[56:32] | type=2
  i32x8 g1;
  g1[0] = (int)(1u << 16);                                // mask=0, data_size=1 (2 bytes)
  g1[1] = (int)((stride_elems & 0xFFFFu) << 16);          // tensor_dim0[15:0] @ bits 63:48
  g1[2] = (int)(((stride_elems >> 16) & 0xFFFFu) | ((nrows & 0xFFFFu) << 16));
  g1[3] = (int)(((nrows >> 16) & 0xFFFFu) | (tile_w << 16)); // tile_dim0 @ bits 127:112
  g1[4] = (int)(tile_h & 0xFFFFu);                        // tile_dim1; tile_dim2=0
  g1[5] = (int)stride_elems;                              // tensor_dim0_stride lo32
  g1[6] = 0;                                              // stride hi16 | dim1_stride lo16
  g1[7] = 0;
  i32x4 z4 = {};
  i32x8 z8 = {};
  // 6-arg form (clang-23 / therock-10.0 headers): (g0, g1, g2, g3, extra, cpol)
  __builtin_amdgcn_tensor_load_to_lds(g0, g1, z4, z4, z8, 0);
}
#endif

// ---------------- elementwise f32 -> bf16 ----------------
__global__ void __launch_bounds__(256) cvt_bf16_kernel(const float* __restrict__ src,
                                                       unsigned short* __restrict__ dst, int n) {
  int i = blockIdx.x * 256 + threadIdx.x;
  if (i < n) dst[i] = f2bf(src[i]);
}

// ------------- W (KxN f32, row-major) -> WT (NxK bf16, row-major) -------------
__global__ void __launch_bounds__(256) wt_transpose_kernel(const float* __restrict__ W,
                                                           unsigned short* __restrict__ WT,
                                                           int K, int N) {
  int i = blockIdx.x * 256 + threadIdx.x;
  if (i >= K * N) return;
  int n = i / K, k = i - n * K;
  WT[(size_t)n * K + k] = f2bf(W[(size_t)k * N + n]);
}

// ---------------- bf16 WMMA GEMM: C(MxN f32) = A(MxK) * BT(NxK)^T + bias ----------------
// Block tile 64(M) x 256(N); 8 waves, each 32x64. K staged in 32-wide double-buffered
// LDS tiles filled by the Tensor Data Mover (wave 0 issues, TENSORcnt + barrier syncs).
__global__ void __launch_bounds__(256) gemm_bf16_kernel(const unsigned short* __restrict__ A,
                                                        const unsigned short* __restrict__ BT,
                                                        const float* __restrict__ bias,
                                                        float* __restrict__ C,
                                                        int M, int N, int K) {
  const int lane  = threadIdx.x & 31;
  const int wave  = threadIdx.x >> 5;
  const int lhalf = lane >> 4;
  const int l16   = lane & 15;
  const int mw = (wave >> 2) * 32;     // wave's local M offset within the 64-row block tile
  const int nw = (wave & 3) * 64;      // wave's local N offset within the 256-col block tile
  const int mblock = blockIdx.y * 64;
  const int nblock = blockIdx.x * 256;

  v8f acc[2][4];
#pragma unroll
  for (int i = 0; i < 2; ++i)
#pragma unroll
    for (int j = 0; j < 4; ++j) acc[i][j] = {};

#if USE_TDM
  __shared__ unsigned short Asm[2][64 * 32];   //  4 KB per buffer
  __shared__ unsigned short Bsm[2][256 * 32];  // 16 KB per buffer

  if (wave == 0) {
    tdm_load_2d(lds_addr_of(&Asm[0][0]), A  + (size_t)mblock * K, 32, 64,  (unsigned)K, (unsigned)M);
    tdm_load_2d(lds_addr_of(&Bsm[0][0]), BT + (size_t)nblock * K, 32, 256, (unsigned)K, (unsigned)N);
  }
  int cur = 0;
  for (int k0 = 0; k0 < K; k0 += 32) {
    if (wave == 0) __builtin_amdgcn_s_wait_tensorcnt(0);
    __syncthreads();                       // tile `cur` is resident; prev reads of cur^1 done
    if (wave == 0 && k0 + 32 < K) {        // fill next buffer while computing this one
      tdm_load_2d(lds_addr_of(&Asm[cur ^ 1][0]), A  + (size_t)mblock * K + (k0 + 32), 32, 64,  (unsigned)K, (unsigned)M);
      tdm_load_2d(lds_addr_of(&Bsm[cur ^ 1][0]), BT + (size_t)nblock * K + (k0 + 32), 32, 256, (unsigned)K, (unsigned)N);
    }
    const unsigned short* As = &Asm[cur][0];
    const unsigned short* Bs = &Bsm[cur][0];
    v16bf af0 = ld_frag(As + (mw + l16) * 32 + lhalf * 16,
                        As + (mw + l16) * 32 + lhalf * 16 + 8);
    v16bf af1 = ld_frag(As + (mw + 16 + l16) * 32 + lhalf * 16,
                        As + (mw + 16 + l16) * 32 + lhalf * 16 + 8);
#pragma unroll
    for (int j = 0; j < 4; ++j) {
      const unsigned short* bp = Bs + (nw + j * 16 + l16) * 32 + lhalf * 16;
      v16bf bf = ld_frag(bp, bp + 8);
      acc[0][j] = wmma_bf16(af0, bf, acc[0][j]);
      acc[1][j] = wmma_bf16(af1, bf, acc[1][j]);
    }
    cur ^= 1;
  }
#else
  // Fallback: direct-global fragment loads (round-1 path)
  const unsigned short* a0p = A + (size_t)(mblock + mw + l16) * K;
  const unsigned short* a1p = A + (size_t)(mblock + mw + 16 + l16) * K;
  for (int k0 = 0; k0 < K; k0 += 32) {
    if (k0 + 128 < K) { __builtin_prefetch(a0p + k0 + 128); __builtin_prefetch(a1p + k0 + 128); }
    v16bf af0 = ld_frag(a0p + k0 + lhalf * 16, a0p + k0 + lhalf * 16 + 8);
    v16bf af1 = ld_frag(a1p + k0 + lhalf * 16, a1p + k0 + lhalf * 16 + 8);
#pragma unroll
    for (int j = 0; j < 4; ++j) {
      const unsigned short* bp = BT + (size_t)(nblock + nw + j * 16 + l16) * K + k0 + lhalf * 16;
      v16bf bf = ld_frag(bp, bp + 8);
      acc[0][j] = wmma_bf16(af0, bf, acc[0][j]);
      acc[1][j] = wmma_bf16(af1, bf, acc[1][j]);
    }
  }
#endif

#pragma unroll
  for (int i = 0; i < 2; ++i)
#pragma unroll
    for (int j = 0; j < 4; ++j) {
      const int n = nblock + nw + j * 16 + l16;
      const float bv = bias ? bias[n] : 0.0f;
#pragma unroll
      for (int r = 0; r < 8; ++r) {
        const int m = mblock + mw + i * 16 + lhalf * 8 + r;
        C[(size_t)m * N + n] = acc[i][j][r] + bv;
      }
    }
}

// ------- per-head RMSNorm + repack to (B, NH, S, D) bf16; 8 rows per block, wave per row -------
__global__ void __launch_bounds__(256) rmsnorm_pack_kernel(const float* __restrict__ src,   // (B,S,NH,D)
                                                           const float* __restrict__ gamma, // (D)
                                                           unsigned short* __restrict__ dst,// (B,NH,S,D)
                                                           int S, int NH) {
  const int rid  = blockIdx.x * 8 + (threadIdx.x >> 5);
  const int lane = threadIdx.x & 31;
  const int hh = rid % NH;
  int t = rid / NH;
  const int s = t % S;
  const int b = t / S;

  const float* p = src + (size_t)rid * HEAD_DIM + lane * 4;
  float4 xv = *reinterpret_cast<const float4*>(p);
  float ss = xv.x * xv.x + xv.y * xv.y + xv.z * xv.z + xv.w * xv.w;
  ss += __shfl_xor(ss, 1, 32);
  ss += __shfl_xor(ss, 2, 32);
  ss += __shfl_xor(ss, 4, 32);
  ss += __shfl_xor(ss, 8, 32);
  ss += __shfl_xor(ss, 16, 32);
  const float rinv = rsqrtf(ss * (1.0f / HEAD_DIM) + 1e-8f);

  float4 gv = *reinterpret_cast<const float4*>(gamma + lane * 4);
  ushort4 o;
  o.x = f2bf(xv.x * rinv * gv.x);
  o.y = f2bf(xv.y * rinv * gv.y);
  o.z = f2bf(xv.z * rinv * gv.z);
  o.w = f2bf(xv.w * rinv * gv.w);
  unsigned short* q = dst + (((size_t)b * NH + hh) * S + s) * HEAD_DIM + lane * 4;
  *reinterpret_cast<ushort4*>(q) = o;
}

// ------- V (B,S,G,D f32) -> Vt (B,G,D,S bf16) -------
__global__ void __launch_bounds__(256) v_transpose_kernel(const float* __restrict__ v,
                                                          unsigned short* __restrict__ vt, int total) {
  int i = blockIdx.x * 256 + threadIdx.x;
  if (i >= total) return;
  const int d = i & (HEAD_DIM - 1);
  int t = i >> 7;
  const int g = t % NUM_KV_GROUPS; t /= NUM_KV_GROUPS;
  const int s = t % SEQ;
  const int b = t / SEQ;
  vt[(((size_t)b * NUM_KV_GROUPS + g) * HEAD_DIM + d) * SEQ + s] = f2bf(v[i]);
}

// ------- Flash attention: wave owns a 16-row Q tile; keys in 32-wide steps -------
__global__ void __launch_bounds__(256) attn_kernel(const unsigned short* __restrict__ Qb, // (B*H, S, D)
                                                   const unsigned short* __restrict__ Kb, // (B*G, S, D)
                                                   const unsigned short* __restrict__ Vt, // (B*G, D, S)
                                                   unsigned short* __restrict__ Ctx,      // (B, S, H*D) bf16
                                                   int S) {
  __shared__ unsigned short pbuf_all[8 * 16 * 32];
  const int wave  = threadIdx.x >> 5;
  const int lane  = threadIdx.x & 31;
  const int lhalf = lane >> 4;
  const int l16   = lane & 15;
  unsigned short* pbuf = pbuf_all + wave * (16 * 32);

  const int bh = blockIdx.x;
  const int b = bh / N_HEADS, h = bh % N_HEADS;
  const int g = h / GROUP_SIZE;                 // repeat_interleave mapping
  const int q0 = blockIdx.y * 128 + wave * 16;

  const unsigned short* Qp = Qb + (size_t)bh * S * HEAD_DIM;
  const unsigned short* Kp = Kb + (size_t)(b * NUM_KV_GROUPS + g) * S * HEAD_DIM;
  const unsigned short* Vp = Vt + (size_t)(b * NUM_KV_GROUPS + g) * HEAD_DIM * S;

  // Preload Q A-fragments: 4 K-chunks of 32 over D=128
  v16bf qa[4];
  {
    const unsigned short* qr = Qp + (size_t)(q0 + l16) * HEAD_DIM;
#pragma unroll
    for (int c = 0; c < 4; ++c)
      qa[c] = ld_frag(qr + c * 32 + lhalf * 16, qr + c * 32 + lhalf * 16 + 8);
  }

  v8f O[8];
#pragma unroll
  for (int n = 0; n < 8; ++n) O[n] = {};
  float mrow[8], lrow[8];
#pragma unroll
  for (int r = 0; r < 8; ++r) { mrow[r] = -__builtin_inff(); lrow[r] = 0.0f; }

  const float scale = 0.08838834764831845f;     // 1/sqrt(128)
  const int qlast = q0 + 15;

  for (int j = 0; j <= qlast; j += 32) {
    // ---- scores: S0 = Q*K[j..j+15]^T, S1 = Q*K[j+16..j+31]^T ----
    v8f s0 = {}, s1 = {};
#pragma unroll
    for (int c = 0; c < 4; ++c) {
      const unsigned short* k0p = Kp + (size_t)(j + l16) * HEAD_DIM + c * 32 + lhalf * 16;
      v16bf kf0 = ld_frag(k0p, k0p + 8);
      s0 = wmma_bf16(qa[c], kf0, s0);
      const unsigned short* k1p = Kp + (size_t)(j + 16 + l16) * HEAD_DIM + c * 32 + lhalf * 16;
      v16bf kf1 = ld_frag(k1p, k1p + 8);
      s1 = wmma_bf16(qa[c], kf1, s1);
    }
    // ---- causal mask + online softmax (row lives in one 16-lane half, one VGPR) ----
#pragma unroll
    for (int r = 0; r < 8; ++r) {
      const int row = q0 + lhalf * 8 + r;
      const int c0 = j + l16, c1 = j + 16 + l16;
      float x0 = (c0 <= row) ? s0[r] * scale : -__builtin_inff();
      float x1 = (c1 <= row) ? s1[r] * scale : -__builtin_inff();
      float mx = fmaxf(x0, x1);
      mx = fmaxf(mx, __shfl_xor(mx, 1, 32));
      mx = fmaxf(mx, __shfl_xor(mx, 2, 32));
      mx = fmaxf(mx, __shfl_xor(mx, 4, 32));
      mx = fmaxf(mx, __shfl_xor(mx, 8, 32));
      const float mnew = fmaxf(mrow[r], mx);
      const float alpha = __expf(mrow[r] - mnew);
      const float p0 = __expf(x0 - mnew);
      const float p1 = __expf(x1 - mnew);
      float rs = p0 + p1;
      rs += __shfl_xor(rs, 1, 32);
      rs += __shfl_xor(rs, 2, 32);
      rs += __shfl_xor(rs, 4, 32);
      rs += __shfl_xor(rs, 8, 32);
      lrow[r] = lrow[r] * alpha + rs;
      mrow[r] = mnew;
#pragma unroll
      for (int n = 0; n < 8; ++n) O[n][r] *= alpha;
      const int mloc = lhalf * 8 + r;
      pbuf[mloc * 32 + l16]      = f2bf(p0);
      pbuf[mloc * 32 + 16 + l16] = f2bf(p1);
    }
    // within-wave LDS RAW: DS pipe is in-order; fence the compiler + wait dscnt
    asm volatile("s_wait_dscnt 0" ::: "memory");
    // re-gather P (16x32) as an A-fragment
    v16bf pf = ld_frag(&pbuf[l16 * 32 + lhalf * 16], &pbuf[l16 * 32 + lhalf * 16 + 8]);
    // ---- O += P * V[j..j+31]  (B-fragment from Vt rows = d, contiguous key cols) ----
#pragma unroll
    for (int n = 0; n < 8; ++n) {
      const unsigned short* vp = Vp + (size_t)(n * 16 + l16) * S + j + lhalf * 16;
      v16bf vf = ld_frag(vp, vp + 8);
      O[n] = wmma_bf16(pf, vf, O[n]);
    }
  }

  // ---- normalize and write context in (B, S, H*D) bf16 ----
#pragma unroll
  for (int r = 0; r < 8; ++r) lrow[r] = 1.0f / lrow[r];
#pragma unroll
  for (int n = 0; n < 8; ++n) {
#pragma unroll
    for (int r = 0; r < 8; ++r) {
      const int srow = q0 + lhalf * 8 + r;
      const int d = n * 16 + l16;
      Ctx[(size_t)(b * S + srow) * N_EMBD + h * HEAD_DIM + d] = f2bf(O[n][r] * lrow[r]);
    }
  }
}

extern "C" void kernel_launch(void* const* d_in, const int* in_sizes, int n_in,
                              void* d_out, int out_size, void* d_ws, size_t ws_size,
                              hipStream_t stream) {
  (void)in_sizes; (void)n_in; (void)out_size; (void)ws_size;
  const float* x  = (const float*)d_in[0];
  const float* Wq = (const float*)d_in[1];
  const float* bq = (const float*)d_in[2];
  const float* Wk = (const float*)d_in[3];
  const float* bk = (const float*)d_in[4];
  const float* Wv = (const float*)d_in[5];
  const float* bv = (const float*)d_in[6];
  const float* gq = (const float*)d_in[7];
  const float* gk = (const float*)d_in[8];
  const float* Wo = (const float*)d_in[9];
  const float* bo = (const float*)d_in[10];
  float* out = (float*)d_out;

  const int M    = BATCH * SEQ;              // 4096
  const int NKV  = NUM_KV_GROUPS * HEAD_DIM; // 256

  char* ws = (char*)d_ws;
  size_t off = 0;
  auto alloc = [&](size_t bytes) -> void* {
    void* p = ws + off;
    off += (bytes + 255) & ~(size_t)255;
    return p;
  };
  unsigned short* x_bf  = (unsigned short*)alloc((size_t)M * N_EMBD * 2);
  unsigned short* WqT   = (unsigned short*)alloc((size_t)N_EMBD * N_EMBD * 2);
  unsigned short* WkT   = (unsigned short*)alloc((size_t)NKV * N_EMBD * 2);
  unsigned short* WvT   = (unsigned short*)alloc((size_t)NKV * N_EMBD * 2);
  unsigned short* WoT   = (unsigned short*)alloc((size_t)N_EMBD * N_EMBD * 2);
  float*          q_f   = (float*)alloc((size_t)M * N_EMBD * 4);
  float*          k_f   = (float*)alloc((size_t)M * NKV * 4);
  float*          v_f   = (float*)alloc((size_t)M * NKV * 4);
  unsigned short* q_bf  = (unsigned short*)alloc((size_t)M * N_EMBD * 2);
  unsigned short* k_bf  = (unsigned short*)alloc((size_t)M * NKV * 2);
  unsigned short* vt_bf = (unsigned short*)alloc((size_t)M * NKV * 2);
  // context reuses the x_bf slot: x_bf is dead after the QKV GEMMs (stream-ordered)
  unsigned short* ctx_bf = x_bf;

  // 1. x -> bf16
  {
    int n = M * N_EMBD;
    cvt_bf16_kernel<<<(n + 255) / 256, 256, 0, stream>>>(x, x_bf, n);
  }
  // 2. weights -> transposed bf16
  wt_transpose_kernel<<<(N_EMBD * N_EMBD + 255) / 256, 256, 0, stream>>>(Wq, WqT, N_EMBD, N_EMBD);
  wt_transpose_kernel<<<(N_EMBD * NKV + 255) / 256, 256, 0, stream>>>(Wk, WkT, N_EMBD, NKV);
  wt_transpose_kernel<<<(N_EMBD * NKV + 255) / 256, 256, 0, stream>>>(Wv, WvT, N_EMBD, NKV);
  wt_transpose_kernel<<<(N_EMBD * N_EMBD + 255) / 256, 256, 0, stream>>>(Wo, WoT, N_EMBD, N_EMBD);
  // 3. QKV projections (WMMA + TDM-staged LDS tiles)
  gemm_bf16_kernel<<<dim3(N_EMBD / 256, M / 64), 256, 0, stream>>>(x_bf, WqT, bq, q_f, M, N_EMBD, N_EMBD);
  gemm_bf16_kernel<<<dim3(NKV / 256, M / 64), 256, 0, stream>>>(x_bf, WkT, bk, k_f, M, NKV, N_EMBD);
  gemm_bf16_kernel<<<dim3(NKV / 256, M / 64), 256, 0, stream>>>(x_bf, WvT, bv, v_f, M, NKV, N_EMBD);
  // 4. per-head RMSNorm + repack (Q, K); V transpose to (B,G,D,S)
  rmsnorm_pack_kernel<<<(BATCH * SEQ * N_HEADS) / 8, 256, 0, stream>>>(q_f, gq, q_bf, SEQ, N_HEADS);
  rmsnorm_pack_kernel<<<(BATCH * SEQ * NUM_KV_GROUPS) / 8, 256, 0, stream>>>(k_f, gk, k_bf, SEQ, NUM_KV_GROUPS);
  {
    int total = BATCH * SEQ * NUM_KV_GROUPS * HEAD_DIM;
    v_transpose_kernel<<<(total + 255) / 256, 256, 0, stream>>>(v_f, vt_bf, total);
  }
  // 5. flash attention (WMMA) -> context bf16 in (B,S,E)
  attn_kernel<<<dim3(BATCH * N_HEADS, SEQ / 128), 256, 0, stream>>>(q_bf, k_bf, vt_bf, ctx_bf, SEQ);
  // 6. output projection (WMMA) with bias -> f32 out
  gemm_bf16_kernel<<<dim3(N_EMBD / 256, M / 64), 256, 0, stream>>>(ctx_bf, WoT, bo, out, M, N_EMBD, N_EMBD);
}
